// Conv3DFusionModule_39968965656770
// MI455X (gfx1250) — matrix operand order repaired and checked by reading, not compiled, not run on hardware
//
#include <hip/hip_runtime.h>

typedef unsigned short u16;
typedef __attribute__((ext_vector_type(16))) __bf16       v16bf;
typedef __attribute__((ext_vector_type(8)))  float        v8f;
typedef __attribute__((ext_vector_type(8)))  u16          v8u;
typedef __attribute__((ext_vector_type(4)))  unsigned int v4u;

__device__ __forceinline__ u16 f32_to_bf16(float f) {
  unsigned u = __builtin_bit_cast(unsigned, f);
  u += 0x7FFFu + ((u >> 16) & 1u);   // round-to-nearest-even
  return (u16)(u >> 16);
}

__device__ __forceinline__ v16bf frag16(v8u lo, v8u hi) {
  return __builtin_bit_cast(v16bf, __builtin_shufflevector(
      lo, hi, 0, 1, 2, 3, 4, 5, 6, 7, 8, 9, 10, 11, 12, 13, 14, 15));
}

// ---------------- zero the BN accumulators ----------------
__global__ void k_zero(float* __restrict__ p, int n) {
  int i = blockIdx.x * blockDim.x + threadIdx.x;
  if (i < n) p[i] = 0.0f;
}

// ---------------- f32 -> bf16 elementwise ----------------
__global__ void k_cvt(const float* __restrict__ src, u16* __restrict__ dst, int n) {
  int i = blockIdx.x * blockDim.x + threadIdx.x;
  if (i < n) dst[i] = f32_to_bf16(src[i]);
}

// ------- repack weights into WMMA B-fragment order (bf16) -------
// Fragment f = (ko*KS + ks)*NT + nt; element (lane,e) holds
// W[ko][ks*32 + (e&7) + (e>=8 ? 16:0) + (lane>=16 ? 8:0)][nt*16 + (lane&15)]  (ISA 7.12.2 B layout)
__global__ void k_pack(const float* __restrict__ W, u16* __restrict__ dst,
                       int koffs, int Cin, int Cout) {
  int KS = Cin >> 5, NT = Cout >> 4;
  int total = koffs * KS * NT * 512;
  int i = blockIdx.x * blockDim.x + threadIdx.x;
  if (i >= total) return;
  int e    = i & 15;
  int lane = (i >> 4) & 31;
  int f    = i >> 9;
  int nt = f % NT;
  int ks = (f / NT) % KS;
  int ko = f / (NT * KS);
  int k = ks * 32 + (e & 7) + ((e >> 3) << 4) + ((lane >= 16) ? 8 : 0);
  int n = nt * 16 + (lane & 15);
  dst[i] = f32_to_bf16(W[((size_t)ko * Cin + k) * Cout + n]);
}

// ------- unified gather + GEMM tile kernel (conv27 or dense) -------
// Block = 128 threads = 4 waves; 32 rows (2 M-tiles) per block.
// Wave w computes N-tiles {w, w+4}; each B fragment is reused for 2 WMMAs.
// Gather for offset ko+1 is prefetched into registers during compute of ko.
template <int CIN, int COUT, int KOFF>
__global__ __launch_bounds__(128) void k_gemm(
    const u16* __restrict__ x,    // [N, CIN] bf16 bits
    const u16* __restrict__ wpk,  // packed B fragments
    const int* __restrict__ nbr,  // [N,27] (conv) or unused (dense)
    float* __restrict__ y) {      // [N, COUT] f32 pre-BN
  constexpr int KS   = CIN / 32;   // K steps of 32
  constexpr int NT   = COUT / 16;  // N tiles of 16
  constexpr int ROWS = 32;         // 2 M-tiles per block
  constexpr int CPR  = CIN / 8;    // 16-byte chunks per row
  constexpr int PT   = (ROWS * CPR) / 128;  // chunks per thread (2/3/4)

  __shared__ alignas(16) u16 lA[ROWS * CIN];

  const int tid  = threadIdx.x;
  const int lane = tid & 31;
  const int wave = tid >> 5;
  const int hi   = lane >> 4;
  const int mrow = lane & 15;
  const int base = blockIdx.x * ROWS;

  v8f acc[2][2];
#pragma unroll
  for (int a = 0; a < 2; ++a)
#pragma unroll
    for (int b = 0; b < 2; ++b)
#pragma unroll
      for (int r = 0; r < 8; ++r) acc[a][b][r] = 0.0f;

  // register staging for the gathered rows of the current/next offset
  v4u st[PT];

  auto loadChunks = [&](int ko) {
#pragma unroll
    for (int p = 0; p < PT; ++p) {
      int i = tid + p * 128;
      int m = i / CPR, c = i - m * CPR;
      int row;
      if (KOFF == 27) row = nbr[(size_t)(base + m) * 27 + ko];
      else            row = base + m;
      v4u v = {0u, 0u, 0u, 0u};
      if (row >= 0) v = ((const v4u*)(x + (size_t)row * CIN))[c];
      st[p] = v;
    }
  };

  loadChunks(0);

  for (int ko = 0; ko < KOFF; ++ko) {
    __syncthreads();  // all reads of lA from previous offset are done
#pragma unroll
    for (int p = 0; p < PT; ++p) ((v4u*)lA)[tid + p * 128] = st[p];
    __syncthreads();  // lA ready for everyone
    if (ko + 1 < KOFF) loadChunks(ko + 1);  // prefetch next gather during compute

    const u16* wko = wpk + (size_t)ko * (KS * NT * 512);
#pragma unroll
    for (int ks = 0; ks < KS; ++ks) {
      // A fragments (16-bit A 16x32 layout): K = (e<8?e:e+8) + hi*8 + ks*32
      v16bf a[2];
#pragma unroll
      for (int mt = 0; mt < 2; ++mt) {
        const u16* ap = lA + (mt * 16 + mrow) * CIN + ks * 32 + hi * 8;
        a[mt] = frag16(*(const v8u*)ap, *(const v8u*)(ap + 16));
      }
#pragma unroll
      for (int t = 0; t < 2; ++t) {
        const int nt = wave + 4 * t;
        if (nt < NT) {  // constant-folds for NT==4/8; wave-uniform for NT==6
          const u16* bp = wko + ((size_t)ks * NT + nt) * 512 + lane * 16;
          v16bf b = frag16(*(const v8u*)bp, *(const v8u*)(bp + 8));
          acc[0][t] = __builtin_amdgcn_wmma_f32_16x16x32_bf16(
              false, a[0], false, b, (short)0, acc[0][t], false, false);
          acc[1][t] = __builtin_amdgcn_wmma_f32_16x16x32_bf16(
              false, a[1], false, b, (short)0, acc[1][t], false, false);
        }
      }
    }
  }

  // C/D layout: lane holds column c = nt*16 + (lane&15); VGPR r -> M = r + hi*8
#pragma unroll
  for (int t = 0; t < 2; ++t) {
    const int nt = wave + 4 * t;
    if (nt < NT) {
      const int c = nt * 16 + mrow;
#pragma unroll
      for (int mt = 0; mt < 2; ++mt)
#pragma unroll
        for (int r = 0; r < 8; ++r)
          y[(size_t)(base + mt * 16 + r + hi * 8) * COUT + c] = acc[mt][t][r];
    }
  }
}

// ---------------- BN statistics: per-channel sum / sumsq ----------------
__global__ void k_stats(const float* __restrict__ y, float* __restrict__ sums,
                        int N, int Cout) {
  int c = threadIdx.x;  // blockDim.x == Cout (64/96/128)
  float s = 0.f, s2 = 0.f;
  for (int r = blockIdx.x; r < N; r += gridDim.x) {
    float v = y[(size_t)r * Cout + c];
    s += v;
    s2 += v * v;
  }
  atomicAdd(&sums[c], s);
  atomicAdd(&sums[Cout + c], s2);
}

// ---------------- BN finalize: scale/shift per channel ----------------
__global__ void k_finalize(const float* __restrict__ sums,
                           const float* __restrict__ g, const float* __restrict__ b,
                           float* __restrict__ ss, int N, int Cout) {
  int c = threadIdx.x;
  if (c >= Cout) return;
  float inv  = 1.0f / (float)N;
  float mean = sums[c] * inv;
  float var  = sums[Cout + c] * inv - mean * mean;  // biased variance
  float sc   = g[c] * rsqrtf(var + 1e-5f);
  ss[c]        = sc;
  ss[Cout + c] = b[c] - mean * sc;
}

// ---------------- BN apply + ReLU -> bf16 next input / f32 output --------
__global__ void k_apply(const float* __restrict__ y, const float* __restrict__ ss,
                        u16* __restrict__ xnext, float* __restrict__ fout,
                        int total, int Cout) {
  int i = blockIdx.x * blockDim.x + threadIdx.x;
  if (i >= total) return;
  int c = i % Cout;
  float v = fmaf(y[i], ss[c], ss[Cout + c]);
  v = v > 0.f ? v : 0.f;
  if (xnext) xnext[i] = f32_to_bf16(v);
  if (fout)  fout[i]  = v;
}

static void launch_gemm(int layer, const u16* x, const u16* w, const int* nbr,
                        float* y, int N, hipStream_t s) {
  dim3 g(N / 32), b(128);
  switch (layer) {
    case 0: k_gemm< 96,  64, 27><<<g, b, 0, s>>>(x, w, nbr,     y); break;
    case 1: k_gemm< 64,  64,  1><<<g, b, 0, s>>>(x, w, nullptr, y); break;
    case 2: k_gemm< 64,  64, 27><<<g, b, 0, s>>>(x, w, nbr,     y); break;
    case 3: k_gemm< 64,  64,  1><<<g, b, 0, s>>>(x, w, nullptr, y); break;
    case 4: k_gemm< 64,  96,  1><<<g, b, 0, s>>>(x, w, nullptr, y); break;
    case 5: k_gemm< 96, 128, 27><<<g, b, 0, s>>>(x, w, nbr,     y); break;
    case 6: k_gemm<128, 128,  1><<<g, b, 0, s>>>(x, w, nullptr, y); break;
    case 7: k_gemm<128, 128,  1><<<g, b, 0, s>>>(x, w, nullptr, y); break;
  }
}

extern "C" void kernel_launch(void* const* d_in, const int* in_sizes, int n_in,
                              void* d_out, int out_size, void* d_ws, size_t ws_size,
                              hipStream_t stream) {
  // d_in flat order: feat3d(0), feat2d(1), params 24 tensors (2..25), nbr(26)
  const float* feat3d = (const float*)d_in[0];
  const int*   nbr    = (const int*)d_in[26];
  const int    N      = in_sizes[0] / 96;

  struct L { int koff, Cin, Cout, wi; };
  const L Ls[8] = {
      {27,  96,  64,  2},   // conv3 96->64   (W3d0, g0, b0)
      { 1,  64,  64,  5},   // 1x1  64->64    (W3d1)
      {27,  64,  64,  8},   // conv3 64->64   (W3d2)
      { 1,  64,  64, 11},   // 1x1  64->64    (W3d3)
      { 1,  64,  96, 14},   // 1x1  64->96    (We)
      {27,  96, 128, 17},   // conv3 96->128  (Wf0)
      { 1, 128, 128, 20},   // 1x1 128->128   (Wf1)
      { 1, 128, 128, 23},   // 1x1 128->128   (Wf2)
  };

  // ---- workspace layout ----
  char* ws = (char*)d_ws;
  const size_t szX = (size_t)N * 128 * 2;   // bf16 activation buffer (max C)
  const size_t szY = (size_t)N * 128 * 4;   // f32 pre-BN buffer
  u16*   xA = (u16*)ws;
  u16*   xB = (u16*)(ws + szX);
  float* y  = (float*)(ws + 2 * szX);
  char*  wp = ws + 2 * szX + szY;
  u16* wpk[8];
  size_t o = 0;
  for (int i = 0; i < 8; ++i) {
    wpk[i] = (u16*)(wp + o);
    o += (size_t)Ls[i].koff * Ls[i].Cin * Ls[i].Cout * 2;
    o = (o + 255) & ~(size_t)255;
  }
  float* sums = (float*)(wp + o);     // 8 layers x 256 floats (sum | sumsq)
  float* ss   = sums + 8 * 256;       // 8 layers x 256 floats (scale | shift)

  // 1) zero BN accumulators (fresh every call; deterministic under replay)
  k_zero<<<8, 256, 0, stream>>>(sums, 8 * 256);

  // 2) input f32 -> bf16
  {
    int tot = N * 96;
    k_cvt<<<(tot + 255) / 256, 256, 0, stream>>>(feat3d, xA, tot);
  }

  // 3) repack all weights into WMMA B-fragment bf16 layout
  for (int i = 0; i < 8; ++i) {
    int tot = Ls[i].koff * Ls[i].Cin * Ls[i].Cout;
    k_pack<<<(tot + 255) / 256, 256, 0, stream>>>(
        (const float*)d_in[Ls[i].wi], wpk[i], Ls[i].koff, Ls[i].Cin, Ls[i].Cout);
  }

  // 4) layer pipeline: GEMM -> BN stats -> finalize -> apply(+ReLU)
  u16* xcur = xA;
  u16* xnxt = xB;
  for (int i = 0; i < 8; ++i) {
    const L& l = Ls[i];
    launch_gemm(i, xcur, wpk[i], nbr, y, N, stream);
    k_stats<<<512, l.Cout, 0, stream>>>(y, sums + i * 256, N, l.Cout);
    k_finalize<<<1, 128, 0, stream>>>(sums + i * 256,
                                      (const float*)d_in[l.wi + 1],
                                      (const float*)d_in[l.wi + 2],
                                      ss + i * 256, N, l.Cout);
    int tot = N * l.Cout;
    k_apply<<<(tot + 255) / 256, 256, 0, stream>>>(
        y, ss + i * 256,
        (i == 7) ? nullptr : xnxt,
        (i == 7) ? (float*)d_out : nullptr,
        tot, l.Cout);
    u16* t = xcur; xcur = xnxt; xnxt = t;
  }
}
